// Gemma4AudioLayer_33646773797303
// MI455X (gfx1250) — matrix operand order repaired and checked by reading, not compiled, not run on hardware
//
#include <hip/hip_runtime.h>
#include <hip/hip_bf16.h>
#include <math.h>

// ---------------- problem constants ----------------
#define NHEADS 12
#define HD     128
#define CHUNK  128
#define PAST   127
#define CTX    255
#define CTXP   256
#define NBLK   32          // SEQ / CHUNK
#define BATCH  4
#define SEQ    4096
#define HDIM   1536        // NHEADS * HD
#define MROWS  (BATCH * SEQ)

#define Q_SCALE_C 0.1275174f   // (1/sqrt(128)) / ln(2)
#define K_SCALE_C 1.8946355f   // log1p(e) / ln(2)

// ---------------- WMMA plumbing ----------------
typedef __attribute__((ext_vector_type(16))) __bf16 v16bf;
typedef __attribute__((ext_vector_type(8)))  float  v8f;

__device__ __forceinline__ unsigned short f2bf(float x) {
  unsigned u = __float_as_uint(x);
  return (unsigned short)((u + 0x7FFFu + ((u >> 16) & 1u)) >> 16);
}

// A fragment: 16(M) x 32(K) tile from LDS row-major [rows][ldk] bf16.
// Lane layout (ISA 16-bit A 16x32): per lane two contiguous 16B chunks.
__device__ __forceinline__ v16bf frag_a(const unsigned short* lds, int ldk, int m0, int k0, int lane) {
  const int r = lane & 15, half = lane >> 4;
  const int e0 = (m0 + r) * ldk + k0 + 8 * half;
  union { v16bf v; uint4 q[2]; } f;
  f.q[0] = *(const uint4*)&lds[e0];        // K = 8*half + 0..7
  f.q[1] = *(const uint4*)&lds[e0 + 16];   // K = 16 + 8*half + 0..7
  return f.v;
}

// B fragment: 32(K) x 16(N); LDS holds Bt[n][k] row-major [N][ldk] (columns of B contiguous).
// Per lane two 16B chunks: K = 8*half + 0..7 and K = 16 + 8*half + 0..7 (matches frag_a).
__device__ __forceinline__ v16bf frag_b(const unsigned short* lds, int ldk, int n0, int k0, int lane) {
  const int n = lane & 15, half = lane >> 4;
  const int e0 = (n0 + n) * ldk + k0 + 8 * half;
  union { v16bf v; uint4 q[2]; } f;
  f.q[0] = *(const uint4*)&lds[e0];
  f.q[1] = *(const uint4*)&lds[e0 + 16];
  return f.v;
}

__device__ __forceinline__ v8f wmma_bf16(v16bf a, v16bf b, v8f c) {
  return __builtin_amdgcn_wmma_f32_16x16x32_bf16(false, a, false, b, (short)0, c, false, false);
}

// ---------------- CDNA5 async global->LDS staging ----------------
__device__ __forceinline__ void async_b128(void* lds_dst, const void* gsrc) {
  unsigned loff = (unsigned)(unsigned long long)lds_dst;   // low 32 bits = LDS byte offset
  unsigned long long ga = (unsigned long long)gsrc;
  asm volatile("global_load_async_to_lds_b128 %0, %1, off"
               :: "v"(loff), "v"(ga) : "memory");
}
__device__ __forceinline__ void async_wait0() {
  asm volatile("s_wait_asynccnt 0" ::: "memory");
}

// ---------------- fp32 -> bf16 conversion ----------------
__global__ __launch_bounds__(256) void k_f2bf(const float* __restrict__ src,
                                              unsigned short* __restrict__ dst, long long n) {
  long long i = (long long)blockIdx.x * 256 + threadIdx.x;
  if (i < n) dst[i] = f2bf(src[i]);
}

__global__ __launch_bounds__(256) void k_f2bf_pad(const float* __restrict__ src,
                                                  unsigned short* __restrict__ dst,
                                                  int rows_src, int cols, long long n) {
  long long i = (long long)blockIdx.x * 256 + threadIdx.x;
  if (i >= n) return;
  int r = (int)(i / cols);
  dst[i] = (r < rows_src) ? f2bf(src[i]) : (unsigned short)0;
}

// ---------------- generic GEMM: D[m,n] = sum_k A[m,k] * W[n,k] ----------------
// Software-pipelined: async DMA of K-tile (i+1) overlaps WMMA compute of K-tile i.
// mode 0: bf16 out;  1: bf16 * Q_SCALE*softplus(pds[n%128]);  2: bf16 * K_SCALE;  3: f32 out
#define GKC 64
__global__ __launch_bounds__(256) void k_gemm(const unsigned short* __restrict__ A,
                                              const unsigned short* __restrict__ W,
                                              void* __restrict__ Out,
                                              const float* __restrict__ pds,
                                              int N, int K, int mode) {
  __shared__ unsigned short LA[2][128 * GKC];
  __shared__ unsigned short LB[2][128 * GKC];
  const int tid = threadIdx.x, lane = tid & 31, wid = tid >> 5;
  const int wm = wid & 1, wn = wid >> 1;           // 2 x 4 wave grid -> 128x128 tile
  const int mBase = blockIdx.x * 128, nBase = blockIdx.y * 128;
  const int half = lane >> 4, nn = lane & 15;

  auto stage = [&](int buf, int k0) {
#pragma unroll
    for (int t = 0; t < 4; ++t) {
      const int e = (tid + t * 256) * 8;
      const int row = e >> 6, col = e & 63;
      async_b128(&LA[buf][e], &A[(long long)(mBase + row) * K + k0 + col]);
    }
#pragma unroll
    for (int t = 0; t < 4; ++t) {
      const int e = (tid + t * 256) * 8;
      const int row = e >> 6, col = e & 63;
      async_b128(&LB[buf][e], &W[(long long)(nBase + row) * K + k0 + col]);
    }
  };

  v8f acc[4][2] = {};
  int buf = 0;
  stage(0, 0);

  for (int k0 = 0; k0 < K; k0 += GKC) {
    async_wait0();            // tile k0 resident (next tile not yet issued)
    __syncthreads();          // all waves done with the buffer we are about to refill
    if (k0 + GKC < K) {
      stage(buf ^ 1, k0 + GKC);               // DMA next tile during compute
      __builtin_prefetch(&A[(long long)(mBase + (tid >> 1)) * K + k0 + 2 * GKC], 0, 1);
      __builtin_prefetch(&W[(long long)(nBase + (tid >> 1)) * K + k0 + 2 * GKC], 0, 1);
    }
#pragma unroll
    for (int ks = 0; ks < GKC; ks += 32) {
      v16bf bf0 = frag_b(LB[buf], GKC, wn * 32 + 0,  ks, lane);
      v16bf bf1 = frag_b(LB[buf], GKC, wn * 32 + 16, ks, lane);
#pragma unroll
      for (int mt = 0; mt < 4; ++mt) {
        v16bf af = frag_a(LA[buf], GKC, wm * 64 + mt * 16, ks, lane);
        acc[mt][0] = wmma_bf16(af, bf0, acc[mt][0]);
        acc[mt][1] = wmma_bf16(af, bf1, acc[mt][1]);
      }
    }
    buf ^= 1;
  }

#pragma unroll
  for (int mt = 0; mt < 4; ++mt)
#pragma unroll
    for (int nt = 0; nt < 2; ++nt)
#pragma unroll
      for (int i = 0; i < 8; ++i) {
        const int m = mBase + wm * 64 + mt * 16 + half * 8 + i;
        const int n = nBase + wn * 32 + nt * 16 + nn;
        float v = acc[mt][nt][i];
        if (mode == 1) { int d = n & (HD - 1); v *= Q_SCALE_C * log1pf(expf(pds[d])); }
        else if (mode == 2) v *= K_SCALE_C;
        if (mode == 3)
          ((float*)Out)[(long long)m * N + n] = v;
        else
          ((unsigned short*)Out)[(long long)m * N + n] = f2bf(v);
      }
}

// ---------------- per-(b,h,chunk) score kernel: rel-shifted BD + AC ----------------
__global__ __launch_bounds__(256) void k_scores(const unsigned short* __restrict__ Qb,
                                                const unsigned short* __restrict__ Kb,
                                                const unsigned short* __restrict__ Rb,
                                                float* __restrict__ Scores) {
  __shared__ unsigned short LA[128 * 128];   // q block
  __shared__ unsigned short LB[128 * 128];   // rel_k / k_ctx half
  const int nb = blockIdx.x, h = blockIdx.y, b = blockIdx.z;
  const int tid = threadIdx.x, lane = tid & 31, wid = tid >> 5;
  const int wm = wid & 1, wn = wid >> 1;
  const int half = lane >> 4, nn = lane & 15;

  // async-stage Q [128][128]
  {
    const long long rowBase = (long long)b * SEQ + nb * CHUNK;
#pragma unroll
    for (int t = 0; t < 8; ++t) {
      const int e = (tid + t * 256) * 8;
      const int row = e >> 7, col = e & 127;
      async_b128(&LA[e], &Qb[(rowBase + row) * HDIM + h * HD + col]);
    }
  }
  async_wait0();
  __syncthreads();

  const long long sbase = ((((long long)b * NHEADS + h) * NBLK + nb) * CHUNK) * CTXP;

  // ---- BD = q x rel_k^T, scattered with XL rel-shift ----
  for (int ph = 0; ph < 2; ++ph) {
#pragma unroll
    for (int t = 0; t < 8; ++t) {
      const int e = (tid + t * 256) * 8;
      const int row = e >> 7, col = e & 127;
      async_b128(&LB[e], &Rb[(long long)(ph * 128 + row) * HDIM + h * HD + col]);
    }
    async_wait0();
    __syncthreads();
    v8f acc[4][2] = {};
#pragma unroll
    for (int ks = 0; ks < 128; ks += 32) {
      v16bf bf0 = frag_b(LB, 128, wn * 32 + 0,  ks, lane);
      v16bf bf1 = frag_b(LB, 128, wn * 32 + 16, ks, lane);
#pragma unroll
      for (int mt = 0; mt < 4; ++mt) {
        v16bf af = frag_a(LA, 128, wm * 64 + mt * 16, ks, lane);
        acc[mt][0] = wmma_bf16(af, bf0, acc[mt][0]);
        acc[mt][1] = wmma_bf16(af, bf1, acc[mt][1]);
      }
    }
#pragma unroll
    for (int mt = 0; mt < 4; ++mt)
#pragma unroll
      for (int nt = 0; nt < 2; ++nt)
#pragma unroll
        for (int i = 0; i < 8; ++i) {
          const int r = wm * 64 + mt * 16 + half * 8 + i;     // bd row
          const int j = ph * 128 + wn * 32 + nt * 16 + nn;    // rel-pos index
          if (j < CTX) {
            int c = r, x = r + j;
            if (x >= CTX) { c = r + 1; x -= CTX; }
            if (c < CHUNK) Scores[sbase + (long long)c * CTXP + x] = acc[mt][nt][i];
          }
        }
    __threadfence();
    __syncthreads();
  }

  // ---- AC = q x k_ctx^T, accumulate (window zero-padded) ----
  for (int xh = 0; xh < 2; ++xh) {
#pragma unroll
    for (int t = 0; t < 8; ++t) {
      const int e = (tid + t * 256) * 8;
      const int row = e >> 7, col = e & 127;
      const int xg = xh * 128 + row;
      const int pos = nb * CHUNK - PAST + xg;
      if (xg < CTX && pos >= 0 && pos < SEQ)
        async_b128(&LB[e], &Kb[((long long)b * SEQ + pos) * HDIM + h * HD + col]);
      else
        *(uint4*)&LB[e] = make_uint4(0u, 0u, 0u, 0u);
    }
    async_wait0();
    __syncthreads();
    v8f acc[4][2] = {};
#pragma unroll
    for (int ks = 0; ks < 128; ks += 32) {
      v16bf bf0 = frag_b(LB, 128, wn * 32 + 0,  ks, lane);
      v16bf bf1 = frag_b(LB, 128, wn * 32 + 16, ks, lane);
#pragma unroll
      for (int mt = 0; mt < 4; ++mt) {
        v16bf af = frag_a(LA, 128, wm * 64 + mt * 16, ks, lane);
        acc[mt][0] = wmma_bf16(af, bf0, acc[mt][0]);
        acc[mt][1] = wmma_bf16(af, bf1, acc[mt][1]);
      }
    }
#pragma unroll
    for (int mt = 0; mt < 4; ++mt)
#pragma unroll
      for (int nt = 0; nt < 2; ++nt)
#pragma unroll
        for (int i = 0; i < 8; ++i) {
          const int r = wm * 64 + mt * 16 + half * 8 + i;
          const int x = xh * 128 + wn * 32 + nt * 16 + nn;
          const long long o = sbase + (long long)r * CTXP + x;
          Scores[o] += acc[mt][nt][i];   // unique (r,x) per lane-element
        }
    __syncthreads();
  }
}

// ---------------- softcap tanh + softmax over 255 ctx cols (1 wave / row) ----------------
__global__ __launch_bounds__(256) void k_softmax(const float* __restrict__ Scores,
                                                 unsigned short* __restrict__ Wgt,
                                                 long long nrows) {
  const int wid = threadIdx.x >> 5, lane = threadIdx.x & 31;
  const long long row = (long long)blockIdx.x * 8 + wid;
  if (row >= nrows) return;
  const float* s = Scores + row * CTXP;
  float l[8];
  float mx = -1e30f;
#pragma unroll
  for (int t = 0; t < 8; ++t) {
    const int x = t * 32 + lane;
    float v = (x < CTX) ? (50.0f * tanhf(s[x] * 0.02f)) : -1e30f;
    l[t] = v;
    mx = fmaxf(mx, v);
  }
#pragma unroll
  for (int off = 16; off > 0; off >>= 1) mx = fmaxf(mx, __shfl_xor(mx, off, 32));
  float e[8];
  float sum = 0.0f;
#pragma unroll
  for (int t = 0; t < 8; ++t) {
    e[t] = (l[t] > -1e29f) ? exp2f((l[t] - mx) * 1.44269504f) : 0.0f;
    sum += e[t];
  }
#pragma unroll
  for (int off = 16; off > 0; off >>= 1) sum += __shfl_xor(sum, off, 32);
  const float inv = 1.0f / sum;
  unsigned short* w = Wgt + row * CTXP;
#pragma unroll
  for (int t = 0; t < 8; ++t) w[t * 32 + lane] = f2bf(e[t] * inv);  // col 255 -> 0
}

// ---------------- out = w x v_ctx per (b,h,chunk) ----------------
__global__ __launch_bounds__(256) void k_av(const unsigned short* __restrict__ Wgt,
                                            const unsigned short* __restrict__ Vb,
                                            unsigned short* __restrict__ Attn) {
  __shared__ unsigned short LW[128 * 64];    // weight rows x ctx chunk
  __shared__ unsigned short LVT[128 * 64];   // v transposed: [d][ctx chunk]
  const int nb = blockIdx.x, h = blockIdx.y, b = blockIdx.z;
  const int tid = threadIdx.x, lane = tid & 31, wid = tid >> 5;
  const int wm = wid & 1, wn = wid >> 1;
  const int half = lane >> 4, nn = lane & 15;
  const long long wbase = ((((long long)b * NHEADS + h) * NBLK + nb) * CHUNK) * CTXP;

  v8f acc[4][2] = {};
  for (int kc = 0; kc < 4; ++kc) {           // K = ctx in chunks of 64
#pragma unroll
    for (int t = 0; t < 4; ++t) {            // async-stage weight tile
      const int e = (tid + t * 256) * 8;
      const int row = e >> 6, col = e & 63;
      async_b128(&LW[e], &Wgt[wbase + (long long)row * CTXP + kc * 64 + col]);
    }
#pragma unroll
    for (int t = 0; t < 32; ++t) {           // transpose-stage V (needs scalar scatter)
      const int idx = tid + t * 256;
      const int xc = idx >> 7, d = idx & 127;
      const int xg = kc * 64 + xc;
      const int pos = nb * CHUNK - PAST + xg;
      unsigned short v = 0;
      if (xg < CTX && pos >= 0 && pos < SEQ)
        v = Vb[((long long)b * SEQ + pos) * HDIM + h * HD + d];
      LVT[d * 64 + xc] = v;
    }
    async_wait0();
    __syncthreads();
#pragma unroll
    for (int ks = 0; ks < 64; ks += 32) {
      v16bf bf0 = frag_b(LVT, 64, wn * 32 + 0,  ks, lane);
      v16bf bf1 = frag_b(LVT, 64, wn * 32 + 16, ks, lane);
#pragma unroll
      for (int mt = 0; mt < 4; ++mt) {
        v16bf af = frag_a(LW, 64, wm * 64 + mt * 16, ks, lane);
        acc[mt][0] = wmma_bf16(af, bf0, acc[mt][0]);
        acc[mt][1] = wmma_bf16(af, bf1, acc[mt][1]);
      }
    }
    __syncthreads();
  }
#pragma unroll
  for (int mt = 0; mt < 4; ++mt)
#pragma unroll
    for (int nt = 0; nt < 2; ++nt)
#pragma unroll
      for (int i = 0; i < 8; ++i) {
        const int m = wm * 64 + mt * 16 + half * 8 + i;
        const int n = wn * 32 + nt * 16 + nn;
        Attn[((long long)b * SEQ + nb * CHUNK + m) * HDIM + h * HD + n] = f2bf(acc[mt][nt][i]);
      }
}

// ---------------- host side ----------------
extern "C" void kernel_launch(void* const* d_in, const int* in_sizes, int n_in,
                              void* d_out, int out_size, void* d_ws, size_t ws_size,
                              hipStream_t stream) {
  (void)in_sizes; (void)n_in; (void)out_size; (void)ws_size;
  const float* hs    = (const float*)d_in[0];
  const float* pe    = (const float*)d_in[1];
  const float* Wq    = (const float*)d_in[2];
  const float* Wk    = (const float*)d_in[3];
  const float* Wv    = (const float*)d_in[4];
  const float* Wrel  = (const float*)d_in[5];
  const float* Wpost = (const float*)d_in[6];
  const float* pds   = (const float*)d_in[7];

  char* base = (char*)d_ws;
  size_t off = 0;
  auto carve = [&](size_t bytes) -> void* {
    void* p = base + off;
    off += (bytes + 255) & ~(size_t)255;
    return p;
  };
  const size_t nX = (size_t)MROWS * HDIM;
  const size_t nW = (size_t)HDIM * HDIM;
  const size_t nP = (size_t)CTXP * HDIM;

  unsigned short* Xb     = (unsigned short*)carve(nX * 2);
  unsigned short* Wqb    = (unsigned short*)carve(nW * 2);
  unsigned short* Wkb    = (unsigned short*)carve(nW * 2);
  unsigned short* Wvb    = (unsigned short*)carve(nW * 2);
  unsigned short* Wrelb  = (unsigned short*)carve(nW * 2);
  unsigned short* Wpostb = (unsigned short*)carve(nW * 2);
  unsigned short* Pb     = (unsigned short*)carve(nP * 2);
  unsigned short* Qb     = (unsigned short*)carve(nX * 2);
  unsigned short* Kb     = (unsigned short*)carve(nX * 2);
  unsigned short* Vb     = (unsigned short*)carve(nX * 2);
  unsigned short* RelKb  = (unsigned short*)carve(nP * 2);
  const size_t nScores = (size_t)BATCH * NHEADS * NBLK * CHUNK * CTXP;
  float*          Scores = (float*)carve(nScores * 4);
  unsigned short* Wgt    = (unsigned short*)carve(nScores * 2);
  unsigned short* Attn   = (unsigned short*)carve(nX * 2);

  const dim3 blk(256);

  // bf16 conversions
  k_f2bf<<<dim3((unsigned)((nX + 255) / 256)), blk, 0, stream>>>(hs, Xb, (long long)nX);
  k_f2bf<<<dim3((unsigned)((nW + 255) / 256)), blk, 0, stream>>>(Wq, Wqb, (long long)nW);
  k_f2bf<<<dim3((unsigned)((nW + 255) / 256)), blk, 0, stream>>>(Wk, Wkb, (long long)nW);
  k_f2bf<<<dim3((unsigned)((nW + 255) / 256)), blk, 0, stream>>>(Wv, Wvb, (long long)nW);
  k_f2bf<<<dim3((unsigned)((nW + 255) / 256)), blk, 0, stream>>>(Wrel, Wrelb, (long long)nW);
  k_f2bf<<<dim3((unsigned)((nW + 255) / 256)), blk, 0, stream>>>(Wpost, Wpostb, (long long)nW);
  k_f2bf_pad<<<dim3((unsigned)((nP + 255) / 256)), blk, 0, stream>>>(pe, Pb, CTX, HDIM, (long long)nP);

  // projections (fused q/k scaling) + rel_k
  k_gemm<<<dim3(MROWS / 128, HDIM / 128), blk, 0, stream>>>(Xb, Wqb, Qb, pds, HDIM, HDIM, 1);
  k_gemm<<<dim3(MROWS / 128, HDIM / 128), blk, 0, stream>>>(Xb, Wkb, Kb, nullptr, HDIM, HDIM, 2);
  k_gemm<<<dim3(MROWS / 128, HDIM / 128), blk, 0, stream>>>(Xb, Wvb, Vb, nullptr, HDIM, HDIM, 0);
  k_gemm<<<dim3(CTXP / 128, HDIM / 128), blk, 0, stream>>>(Pb, Wrelb, RelKb, nullptr, HDIM, HDIM, 0);

  // logits = shifted BD + AC
  (void)hipMemsetAsync(Scores, 0, nScores * 4, stream);
  k_scores<<<dim3(NBLK, NHEADS, BATCH), blk, 0, stream>>>(Qb, Kb, RelKb, Scores);

  // softcap + softmax
  const long long nrows = (long long)BATCH * NHEADS * NBLK * CHUNK;
  k_softmax<<<dim3((unsigned)(nrows / 8)), blk, 0, stream>>>(Scores, Wgt, nrows);

  // attention output
  k_av<<<dim3(NBLK, NHEADS, BATCH), blk, 0, stream>>>(Wgt, Vb, Attn);

  // final projection -> fp32 output
  k_gemm<<<dim3(MROWS / 128, HDIM / 128), blk, 0, stream>>>(Attn, Wpostb, d_out, nullptr, HDIM, HDIM, 3);
}
